// TemporalSelfAttention_53747220742146
// MI455X (gfx1250) — matrix operand order repaired and checked by reading, not compiled, not run on hardware
//
#include <hip/hip_runtime.h>
#include <hip/hip_bf16.h>
#include <math.h>

// ---------------------------------------------------------------------------
// Temporal self-attention (BEVFormer style) for MI455X / gfx1250.
//   GEMM1: attn_in[N,512] @ [W_off|W_w][512,192]  (f16 WMMA, f32 accum,
//          B double-buffered in LDS via async global->LDS when available)
//   Sample: softmax + bilinear gather from L2-resident feature maps
//   GEMM2: fused[N,256] @ W_out[256,256]          (same WMMA pipeline)
// ---------------------------------------------------------------------------

typedef __attribute__((ext_vector_type(16))) _Float16 v16h;
typedef __attribute__((ext_vector_type(8)))  _Float16 v8h;
typedef __attribute__((ext_vector_type(8)))  float    v8f;
typedef __attribute__((ext_vector_type(4)))  int      v4i;

#define E_DIM     256
#define K1_DIM    512
#define N1_DIM    192   // 128 offset cols + 64 weight-logit cols
#define N2_DIM    256
#define NP        4
#define RADIUS_F  0.15f

#define B1_CHUNK_BYTES (12 * 32 * 32)   // 12 ntiles * 32 lanes * 32B = 12288
#define B2_CHUNK_BYTES (16 * 32 * 32)   // 16 ntiles * 32 lanes * 32B = 16384

// gfx1250 async global->LDS path (ASYNCcnt) if the toolchain exposes it.
#if defined(__has_builtin)
#if __has_builtin(__builtin_amdgcn_global_load_async_to_lds_b128)
#define TSA_ASYNC_LDS 1
#endif
#endif

__device__ __forceinline__ void tsa_stage16(const char* gsrc, char* ldst) {
#ifdef TSA_ASYNC_LDS
    __builtin_amdgcn_global_load_async_to_lds_b128(
        (__attribute__((address_space(1))) v4i*)(gsrc),
        (__attribute__((address_space(3))) v4i*)(ldst), 0, 0);
#else
    *(v4i*)ldst = *(const v4i*)gsrc;
#endif
}

__device__ __forceinline__ void tsa_stage_wait() {
#ifdef TSA_ASYNC_LDS
#if __has_builtin(__builtin_amdgcn_s_wait_asynccnt)
    __builtin_amdgcn_s_wait_asynccnt(0);
#else
    asm volatile("s_wait_asynccnt 0x0" ::: "memory");
#endif
#endif
    __syncthreads();
}

// ---------------------------------------------------------------------------
// Convert attn_in = concat(prev_bev, query) to f16, row-major [N x 512].
// ---------------------------------------------------------------------------
__global__ __launch_bounds__(256) void tsa_convertA(
    const float* __restrict__ prev_bev, const float* __restrict__ query,
    _Float16* __restrict__ Ah, int N)
{
    size_t i = (size_t)blockIdx.x * 256 + threadIdx.x;
    if (i >= (size_t)N * K1_DIM) return;
    int    k = (int)(i & (K1_DIM - 1));
    size_t n = i >> 9;
    float  v = (k < E_DIM) ? prev_bev[n * E_DIM + k] : query[n * E_DIM + (k - E_DIM)];
    Ah[i] = (_Float16)v;
}

// ---------------------------------------------------------------------------
// Pre-pack B into WMMA B-fragment order (ISA 7.12.2, wave32):
// lane L holds column n = L%16, K = (L/16)*8 + [0,8) and 16 + (L/16)*8 + [0,8) of each 32-K tile.
// Packed layout: [ktile][ntile][lane][16 f16]  -> each ktile slice is one
// contiguous chunk so the GEMM stages it into LDS with bulk 16B transfers.
// ---------------------------------------------------------------------------
__global__ __launch_bounds__(256) void tsa_packB1(
    const float* __restrict__ W_off,   // [512 x 128] row-major
    const float* __restrict__ W_w,     // [512 x  64] row-major
    _Float16* __restrict__ Bp1)        // 16*12*32*16 f16
{
    int idx  = blockIdx.x * 256 + threadIdx.x;     // < 98304
    int j    = idx & 15;
    int lane = (idx >> 4) & 31;
    int rest = idx >> 9;
    int t    = rest % 12;
    int kt   = rest / 12;                          // 0..15
    int g    = lane >> 4;
    int ncol = t * 16 + (lane & 15);
    int k    = kt * 32 + g * 8 + (j & 7) + ((j >> 3) << 4);
    float v  = (ncol < 128) ? W_off[k * 128 + ncol] : W_w[k * 64 + (ncol - 128)];
    Bp1[idx] = (_Float16)v;
}

__global__ __launch_bounds__(256) void tsa_packB2(
    const float* __restrict__ W_out,   // [256 x 256] row-major
    _Float16* __restrict__ Bp2)        // 8*16*32*16 f16
{
    int idx  = blockIdx.x * 256 + threadIdx.x;     // < 65536
    int j    = idx & 15;
    int lane = (idx >> 4) & 31;
    int rest = idx >> 9;
    int t    = rest & 15;
    int kt   = rest >> 4;                          // 0..7
    int g    = lane >> 4;
    int ncol = t * 16 + (lane & 15);
    int k    = kt * 32 + g * 8 + (j & 7) + ((j >> 3) << 4);
    Bp2[idx] = (_Float16)W_out[k * N2_DIM + ncol];
}

// ---------------------------------------------------------------------------
// GEMM1: S[N,192] = Ah[N,512] @ B, epilogue: +bias; tanh*R on cols<128.
// 256 threads = 8 waves; each wave owns a 16-row M tile, all 12 N tiles.
// B ktile chunks double-buffered in LDS; next A fragment prefetched.
// ---------------------------------------------------------------------------
__global__ __launch_bounds__(256) void tsa_gemm1(
    const _Float16* __restrict__ Ah, const _Float16* __restrict__ Bp1,
    const float* __restrict__ b_off, const float* __restrict__ b_w,
    float* __restrict__ S, int M)
{
    __shared__ v4i ldsB[2 * B1_CHUNK_BYTES / 16];

    const int tid     = threadIdx.x;
    const int lane    = tid & 31;
    const int wave    = tid >> 5;
    const int g       = lane >> 4;
    const int rowBase = (blockIdx.x * 8 + wave) * 16;
    int arowIdx = rowBase + (lane & 15);
    if (arowIdx >= M) arowIdx = M - 1;                 // clamp; stores guarded
    const _Float16* arow = Ah + (size_t)arowIdx * K1_DIM;

    // stage ktile 0 into buffer 0 (3 x 16B per thread)
    {
        const char* src = (const char*)Bp1;
        char* dst = (char*)ldsB;
#pragma unroll
        for (int i = 0; i < 3; ++i) {
            int o = (tid + i * 256) * 16;
            tsa_stage16(src + o, dst + o);
        }
    }

    v8f acc[12];
#pragma unroll
    for (int t = 0; t < 12; ++t)
#pragma unroll
        for (int i = 0; i < 8; ++i) acc[t][i] = 0.0f;

    v8h a0 = *(const v8h*)(arow + g * 8);
    v8h a1 = *(const v8h*)(arow + 16 + g * 8);

    tsa_stage_wait();

    for (int kt = 0; kt < 16; ++kt) {
        const int cur = kt & 1;
        // stage next ktile into the other buffer (overlaps with WMMA below)
        if (kt + 1 < 16) {
            const char* src = (const char*)Bp1 + (size_t)(kt + 1) * B1_CHUNK_BYTES;
            char* dst = (char*)ldsB + (cur ^ 1) * B1_CHUNK_BYTES;
#pragma unroll
            for (int i = 0; i < 3; ++i) {
                int o = (tid + i * 256) * 16;
                tsa_stage16(src + o, dst + o);
            }
        }
        // prefetch next A fragment
        v8h na0 = a0, na1 = a1;
        if (kt + 1 < 16) {
            const int nk0 = (kt + 1) * 32;
            na0 = *(const v8h*)(arow + nk0 + g * 8);
            na1 = *(const v8h*)(arow + nk0 + 16 + g * 8);
        }

        v16h afrag;
#pragma unroll
        for (int i = 0; i < 8; ++i) { afrag[i] = a0[i]; afrag[8 + i] = a1[i]; }

        const _Float16* bbase = (const _Float16*)((const char*)ldsB + cur * B1_CHUNK_BYTES);
#pragma unroll
        for (int t = 0; t < 12; ++t) {
            v16h bfrag = *(const v16h*)(bbase + ((size_t)(t * 32 + lane)) * 16);
            acc[t] = __builtin_amdgcn_wmma_f32_16x16x32_f16(
                false, afrag, false, bfrag, (short)0, acc[t], false, false);
        }
        a0 = na0; a1 = na1;
        tsa_stage_wait();   // next buffer ready; current safe to overwrite
    }

#pragma unroll
    for (int t = 0; t < 12; ++t) {
        const int col  = t * 16 + (lane & 15);
        const float bias = (col < 128) ? b_off[col] : b_w[col - 128];
#pragma unroll
        for (int r = 0; r < 8; ++r) {
            int row = rowBase + g * 8 + r;
            if (row < M) {
                float v = acc[t][r] + bias;
                if (col < 128) v = tanhf(v) * RADIUS_F;
                S[(size_t)row * N1_DIM + col] = v;
            }
        }
    }
}

// ---------------------------------------------------------------------------
// Sampling: one block per query n, one wave per head (lane = channel).
// Each corner read is a coalesced 128B burst; maps live in L2 (82MB < 192MB).
// Writes fused output directly in f16 for GEMM2.
// ---------------------------------------------------------------------------
__global__ __launch_bounds__(256) void tsa_sample(
    const float* __restrict__ prev_bev,   // [N x 256] == prev map rows
    const float* __restrict__ query,      // [N x 256] == cur map rows
    const float* __restrict__ ref2d,      // [N x 2]
    const float* __restrict__ ego_shift,  // [2]
    const float* __restrict__ S,          // [N x 192] offsets(tanh*R) + logits
    _Float16* __restrict__ fused_h,       // [N x 256]
    const int* __restrict__ bev_h_p, const int* __restrict__ bev_w_p, int N)
{
    const int n = blockIdx.x;
    if (n >= N) return;
    const int h = threadIdx.x >> 5;   // head 0..7
    const int c = threadIdx.x & 31;   // channel in head
    const int H = bev_h_p[0];
    const int W = bev_w_p[0];

    const float* srow = S + (size_t)n * N1_DIM;
    const float rx = ref2d[n * 2 + 0];
    const float ry = ref2d[n * 2 + 1];
    const float ex = ego_shift[0];
    const float ey = ego_shift[1];

    float accum = 0.0f;
#pragma unroll
    for (int s = 0; s < 2; ++s) {
        const float bx = rx + (s == 0 ? ex : 0.0f);
        const float by = ry + (s == 0 ? ey : 0.0f);
        const float* src = (s == 0) ? prev_bev : query;

        // softmax over 4 points of this (stream, head)
        float lg[NP];
        float mx = -INFINITY;
#pragma unroll
        for (int p = 0; p < NP; ++p) {
            lg[p] = srow[128 + s * 32 + h * 4 + p];
            mx = fmaxf(mx, lg[p]);
        }
        float denom = 0.0f;
#pragma unroll
        for (int p = 0; p < NP; ++p) { lg[p] = __expf(lg[p] - mx); denom += lg[p]; }
        const float inv = 1.0f / denom;

#pragma unroll
        for (int p = 0; p < NP; ++p) {
            const float px = bx + srow[s * 64 + h * 8 + p * 2 + 0];
            const float py = by + srow[s * 64 + h * 8 + p * 2 + 1];
            const float ix = px * (float)W - 0.5f;
            const float iy = py * (float)H - 0.5f;
            const float fx = floorf(ix), fy = floorf(iy);
            const int   x0 = (int)fx,   y0 = (int)fy;
            const float wx = ix - fx,   wy = iy - fy;
            float v = 0.0f;
#pragma unroll
            for (int dy = 0; dy < 2; ++dy)
#pragma unroll
                for (int dx = 0; dx < 2; ++dx) {
                    const int xc = x0 + dx, yc = y0 + dy;
                    if (xc >= 0 && xc < W && yc >= 0 && yc < H) {
                        const float wc = (dx ? wx : 1.0f - wx) * (dy ? wy : 1.0f - wy);
                        v += wc * src[((size_t)(yc * W + xc)) * E_DIM + h * 32 + c];
                    }
                }
            accum += v * (lg[p] * inv);
        }
    }
    fused_h[(size_t)n * E_DIM + h * 32 + c] = (_Float16)(accum * 0.5f);
}

// ---------------------------------------------------------------------------
// GEMM2: out[N,256] = fused_h[N,256] @ W_out + b_out   (f32 output)
// Same LDS double-buffered pipeline; 16 N tiles, 8 K steps.
// ---------------------------------------------------------------------------
__global__ __launch_bounds__(256) void tsa_gemm2(
    const _Float16* __restrict__ Fh, const _Float16* __restrict__ Bp2,
    const float* __restrict__ b_out, float* __restrict__ out, int M)
{
    __shared__ v4i ldsB[2 * B2_CHUNK_BYTES / 16];

    const int tid     = threadIdx.x;
    const int lane    = tid & 31;
    const int wave    = tid >> 5;
    const int g       = lane >> 4;
    const int rowBase = (blockIdx.x * 8 + wave) * 16;
    int arowIdx = rowBase + (lane & 15);
    if (arowIdx >= M) arowIdx = M - 1;
    const _Float16* arow = Fh + (size_t)arowIdx * N2_DIM;

    {
        const char* src = (const char*)Bp2;
        char* dst = (char*)ldsB;
#pragma unroll
        for (int i = 0; i < 4; ++i) {
            int o = (tid + i * 256) * 16;
            tsa_stage16(src + o, dst + o);
        }
    }

    v8f acc[16];
#pragma unroll
    for (int t = 0; t < 16; ++t)
#pragma unroll
        for (int i = 0; i < 8; ++i) acc[t][i] = 0.0f;

    v8h a0 = *(const v8h*)(arow + g * 8);
    v8h a1 = *(const v8h*)(arow + 16 + g * 8);

    tsa_stage_wait();

    for (int kt = 0; kt < 8; ++kt) {
        const int cur = kt & 1;
        if (kt + 1 < 8) {
            const char* src = (const char*)Bp2 + (size_t)(kt + 1) * B2_CHUNK_BYTES;
            char* dst = (char*)ldsB + (cur ^ 1) * B2_CHUNK_BYTES;
#pragma unroll
            for (int i = 0; i < 4; ++i) {
                int o = (tid + i * 256) * 16;
                tsa_stage16(src + o, dst + o);
            }
        }
        v8h na0 = a0, na1 = a1;
        if (kt + 1 < 8) {
            const int nk0 = (kt + 1) * 32;
            na0 = *(const v8h*)(arow + nk0 + g * 8);
            na1 = *(const v8h*)(arow + nk0 + 16 + g * 8);
        }

        v16h afrag;
#pragma unroll
        for (int i = 0; i < 8; ++i) { afrag[i] = a0[i]; afrag[8 + i] = a1[i]; }

        const _Float16* bbase = (const _Float16*)((const char*)ldsB + cur * B2_CHUNK_BYTES);
#pragma unroll
        for (int t = 0; t < 16; ++t) {
            v16h bfrag = *(const v16h*)(bbase + ((size_t)(t * 32 + lane)) * 16);
            acc[t] = __builtin_amdgcn_wmma_f32_16x16x32_f16(
                false, afrag, false, bfrag, (short)0, acc[t], false, false);
        }
        a0 = na0; a1 = na1;
        tsa_stage_wait();
    }

#pragma unroll
    for (int t = 0; t < 16; ++t) {
        const int col  = t * 16 + (lane & 15);
        const float bias = b_out[col];
#pragma unroll
        for (int r = 0; r < 8; ++r) {
            int row = rowBase + g * 8 + r;
            if (row < M)
                out[(size_t)row * N2_DIM + col] = acc[t][r] + bias;
        }
    }
}

// ---------------------------------------------------------------------------
extern "C" void kernel_launch(void* const* d_in, const int* in_sizes, int n_in,
                              void* d_out, int out_size, void* d_ws, size_t ws_size,
                              hipStream_t stream) {
    const float* query    = (const float*)d_in[0];
    const float* prev_bev = (const float*)d_in[1];
    const float* ref2d    = (const float*)d_in[2];
    const float* ego      = (const float*)d_in[3];
    const float* W_off    = (const float*)d_in[4];
    const float* b_off    = (const float*)d_in[5];
    const float* W_w      = (const float*)d_in[6];
    const float* b_w      = (const float*)d_in[7];
    const float* W_out    = (const float*)d_in[8];
    const float* b_out    = (const float*)d_in[9];
    const int*   bev_h    = (const int*)d_in[10];
    const int*   bev_w    = (const int*)d_in[11];

    const int N = in_sizes[0] / E_DIM;   // 40000

    // workspace carve-out (256B-aligned slabs)
    char* ws = (char*)d_ws;
    size_t off = 0;
    auto align256 = [](size_t x) { return (x + 255) & ~(size_t)255; };

    _Float16* Ah  = (_Float16*)(ws + off); off += align256((size_t)N * K1_DIM * 2);
    _Float16* Bp1 = (_Float16*)(ws + off); off += align256((size_t)16 * B1_CHUNK_BYTES);
    _Float16* Bp2 = (_Float16*)(ws + off); off += align256((size_t)8  * B2_CHUNK_BYTES);
    float*    S   = (float*)   (ws + off); off += align256((size_t)N * N1_DIM * 4);
    _Float16* Fh  = (_Float16*)(ws + off); off += align256((size_t)N * E_DIM * 2);
    (void)ws_size;

    // 1) precision conversion + weight pre-packing (tiny)
    {
        int blocks = (int)(((size_t)N * K1_DIM + 255) / 256);
        tsa_convertA<<<blocks, 256, 0, stream>>>(prev_bev, query, Ah, N);
    }
    tsa_packB1<<<384, 256, 0, stream>>>(W_off, W_w, Bp1);
    tsa_packB2<<<256, 256, 0, stream>>>(W_out, Bp2);

    // 2) offsets/logits GEMM (WMMA, LDS-staged B) with tanh epilogue
    {
        int blocks = (N + 127) / 128;    // 8 waves x 16 rows per block
        tsa_gemm1<<<blocks, 256, 0, stream>>>(Ah, Bp1, b_off, b_w, S, N);
    }

    // 3) softmax + bilinear sampling + stream fusion -> f16
    tsa_sample<<<N, 256, 0, stream>>>(prev_bev, query, ref2d, ego, S, Fh,
                                      bev_h, bev_w, N);

    // 4) output projection GEMM (WMMA) + bias -> f32 out
    {
        int blocks = (N + 127) / 128;
        tsa_gemm2<<<blocks, 256, 0, stream>>>(Fh, Bp2, b_out, (float*)d_out, N);
    }
}
